// Upsampling_72662256714584
// MI455X (gfx1250) — matrix operand order repaired and checked by reading, not compiled, not run on hardware
//
#include <hip/hip_runtime.h>

// ---------------------------------------------------------------------------
// Fused point-cloud upsampling MLP for MI455X (gfx1250, wave32, WMMA).
//   x_g   = gather(x_down, up_idx)                  [B, Nup, 256]
//   x_lin = leaky(x_g @ W_lin + b_lin, 0.1)         [B, Nup, 128]
//   out   = leaky([x_up, x_lin] @ W_fus + b_fus)    [B, Nup, 128]
// fp32 I/O; GEMMs via bf16x3 split (AhiBhi + AhiBlo + AloBhi) on
// v_wmma_f32_16x16x32_bf16 with fp32 accumulate.
// ---------------------------------------------------------------------------

typedef __attribute__((ext_vector_type(16))) __bf16 v16bf;
typedef __attribute__((ext_vector_type(8)))  float  v8f;

#define BATCH   4
#define NDOWN   16384
#define NUP     65536
#define CDOWN   256
#define CUP     128
#define NTILES  8        // 128 output cols / 16
#define KSTEPS  8        // 256 K / 32
#define WAVES   4
#define BLOCK   128
#define LDSROW  132      // padded row stride (floats) to dodge bank conflicts

__device__ __forceinline__ unsigned short f2bf_rn(float x) {
  union { float f; unsigned int u; } a; a.f = x;
  unsigned int r = a.u + 0x7FFFu + ((a.u >> 16) & 1u);
  return (unsigned short)(r >> 16);
}
__device__ __forceinline__ float bf2f(unsigned short h) {
  union { unsigned int u; float f; } a; a.u = ((unsigned int)h) << 16;
  return a.f;
}

// Split W (fp32, row-major [256,128], K-major) into bf16 hi/lo packed in the
// wave32 B-fragment layout: element (k,n) -> frag(ks=k/32, nt=n/16),
// lane = (n%16) | (((k%32)/8)%2)<<4, elem e = k%8 + 8*((k%32)/16).  Lane's 16 bf16 = 32
// contiguous bytes -> two global_load_b128 at use site.
__global__ void pack_weights_kernel(const float* __restrict__ W,
                                    unsigned short* __restrict__ whi,
                                    unsigned short* __restrict__ wlo) {
  int t = blockIdx.x * blockDim.x + threadIdx.x;
  if (t >= CDOWN * CUP) return;
  int k = t >> 7;          // row (K)
  int n = t & 127;         // col (N)
  float w = W[t];
  unsigned short hi = f2bf_rn(w);
  unsigned short lo = f2bf_rn(w - bf2f(hi));
  int ks = k >> 5, kk = k & 31, hh = (kk >> 3) & 1, e = (kk & 7) | ((kk >> 4) << 3);
  int nt = n >> 4, lane = (n & 15) | (hh << 4);
  int off = ((ks * NTILES + nt) * 32 + lane) * 16 + e;
  whi[off] = hi;
  wlo[off] = lo;
}

// Build the bf16 hi/lo A fragments from 16 fp32 values already in the
// documented per-lane K order.
__device__ __forceinline__ void split_frag(const float (&vals)[16],
                                           v16bf& ahi, v16bf& alo) {
  union { unsigned short u[16]; v16bf v; } H, L;
#pragma unroll
  for (int e = 0; e < 16; ++e) {
    unsigned short hi = f2bf_rn(vals[e]);
    H.u[e] = hi;
    L.u[e] = f2bf_rn(vals[e] - bf2f(hi));
  }
  ahi = H.v;
  alo = L.v;
}

__device__ __forceinline__ v16bf load_bfrag(const unsigned short* __restrict__ p) {
  union { uint4 q[2]; v16bf v; } u;
  const uint4* q = (const uint4*)p;
  u.q[0] = q[0];
  u.q[1] = q[1];
  return u.v;
}

__device__ __forceinline__ void runs_to_vals(float4 f0, float4 f1, float4 f2,
                                             float4 f3, float (&vals)[16]) {
  vals[0] = f0.x;  vals[1] = f0.y;  vals[2] = f0.z;  vals[3] = f0.w;
  vals[4] = f1.x;  vals[5] = f1.y;  vals[6] = f1.z;  vals[7] = f1.w;
  vals[8] = f2.x;  vals[9] = f2.y;  vals[10] = f2.z; vals[11] = f2.w;
  vals[12] = f3.x; vals[13] = f3.y; vals[14] = f3.z; vals[15] = f3.w;
}

#define WMMA_BF16(A, Bm, C)                                                   \
  __builtin_amdgcn_wmma_f32_16x16x32_bf16(false, (A), false, (Bm), (short)0,  \
                                          (C), false, false)

__global__ __launch_bounds__(BLOCK) void fused_upsample_kernel(
    const float* __restrict__ x_down, const float* __restrict__ x_up,
    const int* __restrict__ up_idx,
    const unsigned short* __restrict__ wlin_hi,
    const unsigned short* __restrict__ wlin_lo,
    const float* __restrict__ b_lin,
    const unsigned short* __restrict__ wfus_hi,
    const unsigned short* __restrict__ wfus_lo,
    const float* __restrict__ b_fus, float* __restrict__ out) {
  __shared__ float lds[WAVES][16][LDSROW];  // per-wave x_lin staging tile

  const int wave = threadIdx.x >> 5;
  const int lane = threadIdx.x & 31;
  const int l15  = lane & 15;   // M row (A) or N col (B/C) within tile
  const int h    = lane >> 4;   // K-half selector

  const int tile = blockIdx.x * WAVES + wave;  // 16384 row-tiles total
  const int b    = tile >> 12;                 // 4096 tiles per batch
  const int m0   = (tile & 4095) << 4;

  // Gather source row for this lane's M row (lanes 16..31 mirror 0..15).
  const int gidx = up_idx[(size_t)b * NUP + m0 + l15];
  const float* __restrict__ xg =
      x_down + ((size_t)b * NDOWN + (size_t)gidx) * CDOWN;
  const float* __restrict__ xur = x_up + ((size_t)b * NUP + m0 + l15) * CUP;

  const v8f vzero = {0.f, 0.f, 0.f, 0.f, 0.f, 0.f, 0.f, 0.f};

  // ---------------- stage 1: x_lin = leaky(x_g @ W_lin + b_lin) -----------
  v8f acc[NTILES];
#pragma unroll
  for (int i = 0; i < NTILES; ++i) acc[i] = vzero;

#pragma unroll
  for (int ks = 0; ks < KSTEPS; ++ks) {
    // A-layout (16-bit, 16x32): lane half h -> K runs [32ks+8h, +8) and
    // [32ks+16+8h, +8) -- two contiguous 8-float runs per lane.
    const float* s0 = xg + ks * 32 + 8 * h;
    const float* s1 = s0 + 16;
    float vals[16];
    runs_to_vals(*(const float4*)(s0), *(const float4*)(s0 + 4),
                 *(const float4*)(s1), *(const float4*)(s1 + 4), vals);
    v16bf ahi, alo;
    split_frag(vals, ahi, alo);
#pragma unroll
    for (int nt = 0; nt < NTILES; ++nt) {
      const int off = ((ks * NTILES + nt) * 32 + lane) * 16;
      v16bf bhi = load_bfrag(wlin_hi + off);
      v16bf blo = load_bfrag(wlin_lo + off);
      acc[nt] = WMMA_BF16(ahi, bhi, acc[nt]);
      acc[nt] = WMMA_BF16(ahi, blo, acc[nt]);
      acc[nt] = WMMA_BF16(alo, bhi, acc[nt]);
    }
  }

  // Bias + LeakyReLU; spill x_lin tile to wave-private LDS region.
  // C-layout: VGPR v holds row m = v + 8h, col n = nt*16 + l15.
#pragma unroll
  for (int nt = 0; nt < NTILES; ++nt) {
    const float bl = b_lin[nt * 16 + l15];
#pragma unroll
    for (int v = 0; v < 8; ++v) {
      float x = acc[nt][v] + bl;
      x = (x >= 0.0f) ? x : 0.1f * x;
      lds[wave][v + 8 * h][nt * 16 + l15] = x;
    }
  }
  // Same wave writes & reads its own LDS region; DS ops are in-order and the
  // compiler inserts the required s_wait_dscnt -- no barrier needed.

  // ------- stage 2: out = leaky([x_up, x_lin] @ W_fus + b_fus) ------------
  v8f acc2[NTILES];
#pragma unroll
  for (int i = 0; i < NTILES; ++i) acc2[i] = vzero;

#pragma unroll
  for (int ks = 0; ks < KSTEPS; ++ks) {
    float vals[16];
    if (ks < 4) {  // K = 0..127 <- x_up rows of W_fus
      const float* s0 = xur + ks * 32 + 8 * h;
      const float* s1 = s0 + 16;
      runs_to_vals(*(const float4*)(s0), *(const float4*)(s0 + 4),
                   *(const float4*)(s1), *(const float4*)(s1 + 4), vals);
    } else {       // K = 128..255 <- x_lin (from LDS) rows of W_fus
      const float* s0 = &lds[wave][l15][(ks - 4) * 32 + 8 * h];
      const float* s1 = s0 + 16;
      runs_to_vals(*(const float4*)(s0), *(const float4*)(s0 + 4),
                   *(const float4*)(s1), *(const float4*)(s1 + 4), vals);
    }
    v16bf ahi, alo;
    split_frag(vals, ahi, alo);
#pragma unroll
    for (int nt = 0; nt < NTILES; ++nt) {
      const int off = ((ks * NTILES + nt) * 32 + lane) * 16;
      v16bf bhi = load_bfrag(wfus_hi + off);
      v16bf blo = load_bfrag(wfus_lo + off);
      acc2[nt] = WMMA_BF16(ahi, bhi, acc2[nt]);
      acc2[nt] = WMMA_BF16(ahi, blo, acc2[nt]);
      acc2[nt] = WMMA_BF16(alo, bhi, acc2[nt]);
    }
  }

  // Bias + LeakyReLU + store.
  float* __restrict__ orow = out + ((size_t)b * NUP + m0) * CUP;
#pragma unroll
  for (int nt = 0; nt < NTILES; ++nt) {
    const float bf = b_fus[nt * 16 + l15];
#pragma unroll
    for (int v = 0; v < 8; ++v) {
      float x = acc2[nt][v] + bf;
      x = (x >= 0.0f) ? x : 0.1f * x;
      orow[(size_t)(v + 8 * h) * CUP + nt * 16 + l15] = x;
    }
  }
}

extern "C" void kernel_launch(void* const* d_in, const int* in_sizes, int n_in,
                              void* d_out, int out_size, void* d_ws,
                              size_t ws_size, hipStream_t stream) {
  const float* x_down = (const float*)d_in[0];
  const float* x_up   = (const float*)d_in[1];
  const int*   up_idx = (const int*)d_in[2];  // integer input per harness ABI
  const float* W_lin  = (const float*)d_in[3];
  const float* b_lin  = (const float*)d_in[4];
  const float* W_fus  = (const float*)d_in[5];
  const float* b_fus  = (const float*)d_in[6];
  float* out = (float*)d_out;

  // Workspace: 4 x (256*128) bf16 arrays = 256 KB of packed weight fragments.
  unsigned short* wlin_hi = (unsigned short*)d_ws;
  unsigned short* wlin_lo = wlin_hi + CDOWN * CUP;
  unsigned short* wfus_hi = wlin_lo + CDOWN * CUP;
  unsigned short* wfus_lo = wfus_hi + CDOWN * CUP;

  const int packThreads = CDOWN * CUP;
  pack_weights_kernel<<<(packThreads + 255) / 256, 256, 0, stream>>>(
      W_lin, wlin_hi, wlin_lo);
  pack_weights_kernel<<<(packThreads + 255) / 256, 256, 0, stream>>>(
      W_fus, wfus_hi, wfus_lo);

  const int tiles  = BATCH * (NUP / 16);   // 16384
  const int blocks = tiles / WAVES;        // 4096
  fused_upsample_kernel<<<blocks, BLOCK, 0, stream>>>(
      x_down, x_up, up_idx, wlin_hi, wlin_lo, b_lin, wfus_hi, wfus_lo, b_fus,
      out);
}
